// MRFLoss_67242007987185
// MI455X (gfx1250) — matrix operand order repaired and checked by reading, not compiled, not run on hardware
//
#include <hip/hip_runtime.h>
#include <hip/hip_bf16.h>

typedef _Float16 v16h __attribute__((ext_vector_type(16)));
typedef float    v8f  __attribute__((ext_vector_type(8)));
typedef float    v2f  __attribute__((ext_vector_type(2)));

// ---------------- problem constants ----------------
#define BB   2        // batch
#define CC   256      // channels
#define HH   48
#define WW   48
#define HWSZ 2304     // 48*48
#define HO   46       // output patch grid
#define NP   2116     // 46*46 patches
#define KD   2304     // 256*9 flattened patch dim (36 x 64)
#define MP   2176     // patches padded to multiple of 128
#define GK   2120     // gram K (NP padded to multiple of 4)
#define EPSN 1e-12f
#define LDSTR 72      // LDS row stride in halves (144B = 36 banks, conflict-free b128 frags)
#define PANH  (128 * LDSTR)          // halves per panel buffer
#define PANB  (PANH * 2)             // bytes per panel buffer

// ---------------- ws layout (bytes, all 256-aligned) ----------------
#define SZ_AB   ((size_t)BB * MP * KD * 2)        // 20,054,016
#define OFF_A   ((size_t)0)
#define OFF_B   (OFF_A + SZ_AB)
#define OFF_IP  (OFF_B + SZ_AB)                   // inv-norm pred: B*HW f32
#define OFF_IT  (OFF_IP + (size_t)BB * HWSZ * 4)
#define OFF_BST (OFF_IT + (size_t)BB * HWSZ * 4)  // best: B*MP u64
#define OFF_MPR (OFF_BST + (size_t)BB * MP * 8)   // meanP: B*GK*C f32
#define OFF_MTG (OFF_MPR + (size_t)BB * GK * CC * 4)
#define OFF_SCL (OFF_MTG + (size_t)BB * GK * CC * 4) // 2 f32 accumulators

union FragAB { uint4 u[2]; v16h h; };

static __device__ inline unsigned long long shflx_u64(unsigned long long v, int mask) {
    unsigned lo = (unsigned)v, hi = (unsigned)(v >> 32);
    lo = __shfl_xor(lo, mask, 32);
    hi = __shfl_xor(hi, mask, 32);
    return ((unsigned long long)hi << 32) | lo;
}

// async copy 16B: global -> LDS, tracked by ASYNCcnt (no VGPR data round-trip)
static __device__ inline void async_ld_b128(unsigned ldsOff, const void* g) {
    unsigned long long ga = (unsigned long long)(size_t)g;
    asm volatile("global_load_async_to_lds_b128 %0, %1, off"
                 :: "v"(ldsOff), "v"(ga) : "memory");
}
static __device__ inline void wait_asynccnt0() {
    asm volatile("s_wait_asynccnt 0" ::: "memory");
}

// ---------------- K0: zero init ----------------
__global__ void k0_init(unsigned long long* best, float* meanP, float* meanT, float* scal) {
    int t = blockIdx.x * blockDim.x + threadIdx.x;
    if (t < BB * MP) best[t] = 0ull;
    if (t < 2) scal[t] = 0.f;
    if (t < BB * (GK - NP) * CC) {   // zero gram-K pad rows
        int b = t / ((GK - NP) * CC);
        int r = t % ((GK - NP) * CC);
        size_t o = ((size_t)b * GK + NP + r / CC) * CC + (r % CC);
        meanP[o] = 0.f; meanT[o] = 0.f;
    }
}

// ---------------- K1: per-pixel inverse L2 norm over channels ----------------
__global__ void k1_invnorm(const float* __restrict__ img, float* __restrict__ inv) {
    int t = blockIdx.x * blockDim.x + threadIdx.x;
    if (t >= BB * HWSZ) return;
    int b = t / HWSZ, p = t % HWSZ;
    float ss = 0.f;
    const float* base = img + (size_t)b * CC * HWSZ + p;
    #pragma unroll 4
    for (int c = 0; c < CC; ++c) ss += base[(size_t)c * HWSZ] * base[(size_t)c * HWSZ];
    inv[t] = 1.0f / fmaxf(sqrtf(ss), EPSN);
}

// ---------------- K2: build normalized f16 patch matrix (rows padded with 0) -------
__global__ void k2_fill(const float* __restrict__ img, const float* __restrict__ inv,
                        _Float16* __restrict__ out) {
    long long t = (long long)blockIdx.x * blockDim.x + threadIdx.x;
    if (t >= (long long)BB * MP * KD) return;
    int b = (int)(t / ((long long)MP * KD));
    int rem = (int)(t - (long long)b * MP * KD);
    int m = rem / KD, k = rem % KD;
    float v = 0.f;
    if (m < NP) {
        int i = m / HO, j = m % HO;
        int c = k / 9, r9 = k % 9;
        int pix = (i + r9 / 3) * WW + (j + r9 % 3);
        v = img[((size_t)b * CC + c) * HWSZ + pix] * inv[b * HWSZ + pix];
    }
    out[t] = (_Float16)v;
}

// ---------------- K3: WMMA GEMM sim = A*B^T, double-buffered async LDS staging + argmax
__global__ __launch_bounds__(256)
void k3_gemm_argmax(const _Float16* __restrict__ Am, const _Float16* __restrict__ Bm,
                    unsigned long long* __restrict__ best) {
    __shared__ _Float16 sA[2 * PANH];   // 2 x (128 rows x 64 K-halves, stride 72)
    __shared__ _Float16 sB[2 * PANH];

    const int b    = blockIdx.x;
    const int t    = threadIdx.x;
    const int lane = t & 31;
    const int w    = t >> 5;
    const int r    = lane & 15;
    const int h    = lane >> 4;
    const int mi   = w & 3, ni = w >> 2;            // 4 x 2 waves -> 128 x 128 block tile
    const int mBlk = blockIdx.y * 128;
    const int nBlk = blockIdx.z * 128;
    const int nBase = nBlk + ni * 64;               // this wave: 64 cols
    const int mWave = mi * 32;                      // this wave: rows [mWave, mWave+32)

    // ---- cooperative async-load setup: panel = 128 rows x 64 halves = 1024 x 16B chunks
    const unsigned sAbase = (unsigned)(size_t)sA;
    const unsigned sBbase = (unsigned)(size_t)sB;
    const _Float16* gA[4];
    const _Float16* gB[4];
    unsigned lA[4], lB[4];
    #pragma unroll
    for (int i = 0; i < 4; ++i) {
        int ch  = t + 256 * i;          // 0..1023
        int row = ch >> 3;              // 0..127
        int cp  = ch & 7;               // 16B chunk within 64-half row
        gA[i] = Am + ((size_t)b * MP + mBlk + row) * KD + cp * 8;
        gB[i] = Bm + ((size_t)b * MP + nBlk + row) * KD + cp * 8;
        lA[i] = sAbase + (unsigned)(row * LDSTR + cp * 8) * 2;
        lB[i] = sBbase + (unsigned)(row * LDSTR + cp * 8) * 2;
    }

    v8f acc[2][4];
    #pragma unroll
    for (int f = 0; f < 2; ++f)
        #pragma unroll
        for (int j = 0; j < 4; ++j) acc[f][j] = (v8f)0.0f;

    // issue one panel (8 async b128 per thread) into buffer `buf` for K-base kb
    auto issue_panel = [&](int buf, int kb) {
        const unsigned bo = (unsigned)(buf * PANB);
        #pragma unroll
        for (int i = 0; i < 4; ++i) {
            async_ld_b128(lA[i] + bo, gA[i] + kb);
            async_ld_b128(lB[i] + bo, gB[i] + kb);
        }
        __builtin_prefetch(gA[0] + kb + 64, 0, 3);   // global_prefetch_b8, next panel hint
    };

    // consume one staged panel: two K=32 WMMA sub-steps, 16 wmmas total
    auto compute_panel = [&](const _Float16* cA, const _Float16* cB) {
        #pragma unroll
        for (int sub = 0; sub < 2; ++sub) {
            const int ks = sub * 32;
            FragAB a0, a1, b0, b1, b2, b3;
            // A frag: lane r = row; halves 0..7 -> K=8h+0..7, halves 8..15 -> K=16+8h+0..7
            const _Float16* arow0 = cA + (mWave +      r) * LDSTR + ks + h * 8;
            const _Float16* arow1 = cA + (mWave + 16 + r) * LDSTR + ks + h * 8;
            a0.u[0] = *(const uint4*)(arow0);      a0.u[1] = *(const uint4*)(arow0 + 16);
            a1.u[0] = *(const uint4*)(arow1);      a1.u[1] = *(const uint4*)(arow1 + 16);
            // B frag: lane r = col; halves 0..7 -> K=8h+0..7, halves 8..15 -> K=16+8h+0..7
            const _Float16* bc0 = cB + (ni * 64 +      r) * LDSTR + ks + h * 8;
            const _Float16* bc1 = cB + (ni * 64 + 16 + r) * LDSTR + ks + h * 8;
            const _Float16* bc2 = cB + (ni * 64 + 32 + r) * LDSTR + ks + h * 8;
            const _Float16* bc3 = cB + (ni * 64 + 48 + r) * LDSTR + ks + h * 8;
            b0.u[0] = *(const uint4*)(bc0);        b0.u[1] = *(const uint4*)(bc0 + 16);
            b1.u[0] = *(const uint4*)(bc1);        b1.u[1] = *(const uint4*)(bc1 + 16);
            b2.u[0] = *(const uint4*)(bc2);        b2.u[1] = *(const uint4*)(bc2 + 16);
            b3.u[0] = *(const uint4*)(bc3);        b3.u[1] = *(const uint4*)(bc3 + 16);

            acc[0][0] = __builtin_amdgcn_wmma_f32_16x16x32_f16(false, a0.h, false, b0.h, (short)0, acc[0][0], false, false);
            acc[0][1] = __builtin_amdgcn_wmma_f32_16x16x32_f16(false, a0.h, false, b1.h, (short)0, acc[0][1], false, false);
            acc[0][2] = __builtin_amdgcn_wmma_f32_16x16x32_f16(false, a0.h, false, b2.h, (short)0, acc[0][2], false, false);
            acc[0][3] = __builtin_amdgcn_wmma_f32_16x16x32_f16(false, a0.h, false, b3.h, (short)0, acc[0][3], false, false);
            acc[1][0] = __builtin_amdgcn_wmma_f32_16x16x32_f16(false, a1.h, false, b0.h, (short)0, acc[1][0], false, false);
            acc[1][1] = __builtin_amdgcn_wmma_f32_16x16x32_f16(false, a1.h, false, b1.h, (short)0, acc[1][1], false, false);
            acc[1][2] = __builtin_amdgcn_wmma_f32_16x16x32_f16(false, a1.h, false, b2.h, (short)0, acc[1][2], false, false);
            acc[1][3] = __builtin_amdgcn_wmma_f32_16x16x32_f16(false, a1.h, false, b3.h, (short)0, acc[1][3], false, false);
        }
    };

    // ---- double-buffered pipeline: async DMA of panel k+1 overlaps WMMAs of panel k.
    // Invariant at each wait: only the just-needed panel's asyncs are outstanding.
    // The barrier after the wait also proves all waves finished reading the other
    // buffer (their ds_loads were consumed before they arrived), so overwrite is safe.
    issue_panel(0, 0);
    for (int kb = 0; kb < KD; kb += 128) {           // KD/64 = 36 panels, unroll x2
        wait_asynccnt0();
        __syncthreads();
        if (kb + 64 < KD) issue_panel(1, kb + 64);
        compute_panel(sA, sB);                       // buffer 0

        wait_asynccnt0();
        __syncthreads();
        if (kb + 128 < KD) issue_panel(0, kb + 128);
        compute_panel(sA + PANH, sB + PANH);         // buffer 1
    }

    // Epilogue: per output row M, argmax over the 64 columns this wave owns,
    // then merge across waves/blocks via packed u64 atomicMax.
    // C/D layout: lane r = col offset, half h: VGPR v -> row (v + 8h).
    #pragma unroll
    for (int f = 0; f < 2; ++f) {
        #pragma unroll
        for (int v = 0; v < 8; ++v) {
            float bv = -__builtin_inff();
            int   bn = 0x7FFFFFFF;
            #pragma unroll
            for (int j = 0; j < 4; ++j) {
                int n = nBase + j * 16 + r;
                float val = acc[f][j][v];
                if (n < NP && val > bv) { bv = val; bn = n; }  // strict > keeps smallest n on ties
            }
            unsigned key = __float_as_uint(bv);
            key ^= (key & 0x80000000u) ? 0xFFFFFFFFu : 0x80000000u;  // monotonic map
            unsigned long long packed =
                ((unsigned long long)key << 32) | (unsigned long long)(0xFFFFFFFFu - (unsigned)bn);
            #pragma unroll
            for (int s = 1; s < 16; s <<= 1) {   // reduce across the 16 lanes of this half
                unsigned long long o = shflx_u64(packed, s);
                if (o > packed) packed = o;
            }
            if (r == 0) {
                int row = mBlk + mWave + f * 16 + v + 8 * h;
                atomicMax(best + (size_t)b * MP + row, packed);
            }
        }
    }
}

// ---------------- K4: gather best target patch, patch MSE, per-patch channel means ----
__global__ void k4_gather(const float* __restrict__ pred, const float* __restrict__ targ,
                          const unsigned long long* __restrict__ best,
                          float* __restrict__ meanP, float* __restrict__ meanT,
                          float* __restrict__ scal) {
    int t = blockIdx.x * blockDim.x + threadIdx.x;
    float sse = 0.f;
    if (t < BB * NP * CC) {
        int b = t / (NP * CC);
        int rem = t % (NP * CC);
        int m = rem / CC, c = rem % CC;
        unsigned n = 0xFFFFFFFFu - (unsigned)(best[(size_t)b * MP + m] & 0xFFFFFFFFull);
        if (n >= NP) n = NP - 1;   // safety clamp
        int pi = m / HO, pj = m % HO;
        int ti = (int)n / HO, tj = (int)n % HO;
        const float* pb = pred + ((size_t)b * CC + c) * HWSZ;
        const float* tb = targ + ((size_t)b * CC + c) * HWSZ;
        float sp = 0.f, st = 0.f;
        #pragma unroll
        for (int di = 0; di < 3; ++di)
            #pragma unroll
            for (int dj = 0; dj < 3; ++dj) {
                float pv = pb[(pi + di) * WW + pj + dj];
                float tv = tb[(ti + di) * WW + tj + dj];
                float d = pv - tv;
                sse += d * d; sp += pv; st += tv;
            }
        meanP[((size_t)b * GK + m) * CC + c] = sp * (1.f / 9.f);
        meanT[((size_t)b * GK + m) * CC + c] = st * (1.f / 9.f);
    }
    #pragma unroll
    for (int s = 16; s > 0; s >>= 1) sse += __shfl_xor(sse, s, 32);
    if ((threadIdx.x & 31) == 0 && sse != 0.f) atomicAdd(&scal[0], sse);
}

// ---------------- K5: Gram matrices (x^T x / (c*hw)) + squared diff ----------------
__global__ __launch_bounds__(32)
void k5_gram(const float* __restrict__ meanP, const float* __restrict__ meanT,
             float* __restrict__ scal) {
    const int b = blockIdx.x;
    const int cb = blockIdx.y * 16, db = blockIdx.z * 16;
    const int lane = threadIdx.x & 31;
    const int r = lane & 15, h = lane >> 4;
    const float scale = 1.f / (256.f * 2116.f);
    float local = 0.f;
#if __has_builtin(__builtin_amdgcn_wmma_f32_16x16x4_f32)
    v8f accP = (v8f)0.0f, accT = (v8f)0.0f;
    for (int kb = 0; kb < GK; kb += 4) {
        // A 16x4 f32: lane r = row; VGPR j -> K = kb + 2h + j.  B 4x16: lane r = col; same K map.
        size_t base = ((size_t)b * GK + kb + 2 * h) * CC;
        v2f aP, bP, aT, bT;
        aP[0] = meanP[base + cb + r];      aP[1] = meanP[base + CC + cb + r];
        bP[0] = meanP[base + db + r];      bP[1] = meanP[base + CC + db + r];
        aT[0] = meanT[base + cb + r];      aT[1] = meanT[base + CC + cb + r];
        bT[0] = meanT[base + db + r];      bT[1] = meanT[base + CC + db + r];
        accP = __builtin_amdgcn_wmma_f32_16x16x4_f32(false, aP, false, bP, (short)0, accP, false, false);
        accT = __builtin_amdgcn_wmma_f32_16x16x4_f32(false, aT, false, bT, (short)0, accT, false, false);
    }
    #pragma unroll
    for (int v = 0; v < 8; ++v) {
        float d = (accP[v] - accT[v]) * scale;
        local += d * d;
    }
#else
    // scalar fallback: lane r = col, half h: rows v + 8h
    for (int v = 0; v < 8; ++v) {
        int crow = cb + v + 8 * h, dcol = db + r;
        float dp = 0.f, dt = 0.f;
        for (int m = 0; m < NP; ++m) {
            size_t base = ((size_t)b * GK + m) * CC;
            dp += meanP[base + crow] * meanP[base + dcol];
            dt += meanT[base + crow] * meanT[base + dcol];
        }
        float d = (dp - dt) * scale;
        local += d * d;
    }
#endif
    #pragma unroll
    for (int s = 16; s > 0; s >>= 1) local += __shfl_xor(local, s, 32);
    if (lane == 0) atomicAdd(&scal[1], local);
}

// ---------------- K6: finalize ----------------
__global__ void k6_final(const float* __restrict__ scal, float* __restrict__ out) {
    out[0] = scal[0] * (1.f / ((float)BB * NP * CC * 9.f));
    out[1] = scal[1] * (1.f / ((float)BB * CC * CC));
}

extern "C" void kernel_launch(void* const* d_in, const int* in_sizes, int n_in,
                              void* d_out, int out_size, void* d_ws, size_t ws_size,
                              hipStream_t stream) {
    const float* pred = (const float*)d_in[0];
    const float* targ = (const float*)d_in[1];
    float* out = (float*)d_out;
    char* ws = (char*)d_ws;

    _Float16* Amat = (_Float16*)(ws + OFF_A);
    _Float16* Bmat = (_Float16*)(ws + OFF_B);
    float* invP = (float*)(ws + OFF_IP);
    float* invT = (float*)(ws + OFF_IT);
    unsigned long long* best = (unsigned long long*)(ws + OFF_BST);
    float* meanP = (float*)(ws + OFF_MPR);
    float* meanT = (float*)(ws + OFF_MTG);
    float* scal  = (float*)(ws + OFF_SCL);

    k0_init<<<(BB * MP + 255) / 256, 256, 0, stream>>>(best, meanP, meanT, scal);

    k1_invnorm<<<(BB * HWSZ + 255) / 256, 256, 0, stream>>>(pred, invP);
    k1_invnorm<<<(BB * HWSZ + 255) / 256, 256, 0, stream>>>(targ, invT);

    {
        long long tot = (long long)BB * MP * KD;
        int blocks = (int)((tot + 255) / 256);
        k2_fill<<<blocks, 256, 0, stream>>>(pred, invP, Amat);
        k2_fill<<<blocks, 256, 0, stream>>>(targ, invT, Bmat);
    }

    k3_gemm_argmax<<<dim3(BB, MP / 128, MP / 128), 256, 0, stream>>>(Amat, Bmat, best);

    k4_gather<<<(BB * NP * CC + 255) / 256, 256, 0, stream>>>(pred, targ, best, meanP, meanT, scal);

    k5_gram<<<dim3(BB, CC / 16, CC / 16), 32, 0, stream>>>(meanP, meanT, scal);

    k6_final<<<1, 1, 0, stream>>>(scal, out);
}
